// GTR_40407052320952
// MI455X (gfx1250) — matrix operand order repaired and checked by reading, compiled, not run on hardware
//
#include <hip/hip_runtime.h>
#include <hip/hip_bf16.h>
#include <math.h>

// ---------------------------------------------------------------------------
// GNN TransformerConv x3 + mean pool + fc, for gfx1250 (MI455X).
// Dense linears: v_wmma_f32_16x16x32_f16, wave-per-column-tile with the
// column's B fragments held in registers across all M tiles (no LDS and no
// dscnt waits in the steady-state loop). Bias is pre-splatted into the WMMA
// accumulator; full tiles take an unguarded, EXEC-stable store path.
// Edge softmax via atomics with order-preserving uint max encoding.
// Wave32 throughout.
// ---------------------------------------------------------------------------

typedef __attribute__((ext_vector_type(16))) _Float16 v16h;
typedef __attribute__((ext_vector_type(8)))  _Float16 v8h;
typedef __attribute__((ext_vector_type(8)))  float    v8f;

#define HD    128
#define NHEAD 4
#define DHEAD 32
#define MT_PER_BLK 16               // 16 M-tiles (256 rows) per block
#define ORD_NEG_INF 0x007fffffu     // f2ord(-inf)

static __device__ __forceinline__ unsigned f2ord(float f) {
  unsigned u = __float_as_uint(f);
  return (u & 0x80000000u) ? ~u : (u | 0x80000000u);
}
static __device__ __forceinline__ float ord2f(unsigned u) {
  return __uint_as_float((u & 0x80000000u) ? (u & 0x7fffffffu) : ~u);
}

// ------------------------------- fills -------------------------------------
__global__ void gtr_fill_f32(float* p, float v, long n) {
  long i = (long)blockIdx.x * blockDim.x + threadIdx.x;
  if (i < n) p[i] = v;
}
__global__ void gtr_fill_u32(unsigned* p, unsigned v, long n) {
  long i = (long)blockIdx.x * blockDim.x + threadIdx.x;
  if (i < n) p[i] = v;
}

// ----------------------- f32 -> f16 casts (with K pad) ---------------------
__global__ void gtr_cast_pad(const float* __restrict__ src, _Float16* __restrict__ dst,
                             int Ksrc, int Kdst, long rows) {
  long i = (long)blockIdx.x * blockDim.x + threadIdx.x;
  long total = rows * Kdst;
  if (i >= total) return;
  long r = i / Kdst; int k = (int)(i - r * Kdst);
  dst[i] = (k < Ksrc) ? (_Float16)src[r * Ksrc + k] : (_Float16)0.f;
}
// W is (Ksrc x 128) row-major; pad extra K rows with zero -> (Kdst x 128) f16
__global__ void gtr_castw(const float* __restrict__ W, _Float16* __restrict__ dst,
                          int Ksrc, int Kdst) {
  int i = blockIdx.x * blockDim.x + threadIdx.x;
  if (i >= Kdst * HD) return;
  int k = i / HD;
  dst[i] = (k < Ksrc) ? (_Float16)W[i] : (_Float16)0.f;
}

// --------------------------- WMMA GEMM -------------------------------------
// C[M x 128] = A[M x KD](f16) * W[KD x 128](f16) + bias, f32 out.
// 256 threads = 8 waves. Wave w owns column tile w (cols w*16..w*16+15):
//  - W^T staged once to LDS, then the wave's B fragments for ALL K-steps are
//    pulled into registers (KD/32 x v16h).
//  - Steady-state loop over MT_PER_BLK row tiles: per tile only KD/16 global
//    b128 A-loads + KD/32 WMMAs; no LDS, no dscnt waits, and for full tiles
//    a straight-line 8-store epilogue with no EXEC manipulation.
template <int KD>
__global__ __launch_bounds__(256)
void gtr_gemm_wmma(const _Float16* __restrict__ A, const _Float16* __restrict__ W,
                   const float* __restrict__ bias, float* __restrict__ C, int Nn) {
  constexpr int KSTEPS = KD / 32;
  __shared__ _Float16 Wt[KD * HD];          // Wt[n*KD + k] = W[k*128 + n]
  int tid = threadIdx.x;
  for (int i = tid; i < KD * HD; i += 256) {
    int k = i / HD, n = i - k * HD;
    Wt[n * KD + k] = W[i];
  }
  __syncthreads();

  const int wave  = tid >> 5, lane = tid & 31;
  const int lrow  = lane & 15;
  const int khalf = lane >> 4;              // 0 or 1
  const int col   = wave * 16 + lrow;       // this wave's output column
  const float bcol = bias[col];

  // B fragments for this column tile, all K-steps, held in registers.
  // Lane layout (contiguous k order, same as A): lanes 0-15 -> col, K = ks*32 + 0..15;
  // lanes 16-31 -> col, K = ks*32 + 16..31 (contiguous in LDS-transposed W).
  v16h bfrag[KSTEPS];
#pragma unroll
  for (int ks = 0; ks < KSTEPS; ++ks) {
    const _Float16* bp = &Wt[(size_t)col * KD + ks * 32 + khalf * 16];
    v8h b0 = *(const v8h*)bp;
    v8h b1 = *(const v8h*)(bp + 8);
#pragma unroll
    for (int i = 0; i < 8; ++i) { bfrag[ks][i] = b0[i]; bfrag[ks][i + 8] = b1[i]; }
  }

  const int tile0 = blockIdx.x * MT_PER_BLK;
  for (int t = 0; t < MT_PER_BLK; ++t) {
    int row0 = (tile0 + t) * 16;
    if (row0 >= Nn) break;                  // uniform across block: EXEC full
    int arow = row0 + lrow; if (arow > Nn - 1) arow = Nn - 1;
    const _Float16* Arow = A + (size_t)arow * KD;
    __builtin_prefetch(Arow + 16 * KD, 0, 3);   // next tile, same lane pattern

    // bias pre-splatted into the accumulator: C/D element r is row r of the
    // same output column, so every element gets the same bias.
    v8f acc = (v8f){bcol, bcol, bcol, bcol, bcol, bcol, bcol, bcol};
#pragma unroll
    for (int ks = 0; ks < KSTEPS; ++ks) {
      // A fragment: lane row = lane&15; elems 0..15 = K ks*32+khalf*16..+15
      // (contiguous, matching B fragments)
      v8h a0 = *(const v8h*)(Arow + ks * 32 + khalf * 16);
      v8h a1 = *(const v8h*)(Arow + ks * 32 + khalf * 16 + 8);
      v16h afrag;
#pragma unroll
      for (int i = 0; i < 8; ++i) { afrag[i] = a0[i]; afrag[i + 8] = a1[i]; }
      acc = __builtin_amdgcn_wmma_f32_16x16x32_f16(
          false, afrag, false, bfrag[ks], (short)0, acc, false, false);
    }

    // C/D layout: vgpr r -> row = row0 + khalf*8 + r, col = wave*16+(lane&15)
    float* Crow = C + (size_t)(row0 + khalf * 8) * HD + col;
    if (row0 + 16 <= Nn) {                  // uniform: no EXEC manipulation
#pragma unroll
      for (int r = 0; r < 8; ++r) Crow[(size_t)r * HD] = acc[r];
    } else {                                // at most one partial tail tile
#pragma unroll
      for (int r = 0; r < 8; ++r)
        if (row0 + khalf * 8 + r < Nn) Crow[(size_t)r * HD] = acc[r];
    }
  }
}

// --------------------------- edge phase ------------------------------------
__global__ void gtr_edge_logits(const float* __restrict__ Q, const float* __restrict__ K,
                                const int* __restrict__ src, const int* __restrict__ dst,
                                float* __restrict__ elog, unsigned* __restrict__ mx, long Ne) {
  long idx = (long)blockIdx.x * blockDim.x + threadIdx.x;
  if (idx >= Ne * NHEAD) return;
  long e = idx >> 2; int h = (int)(idx & 3);
  int s = src[e], d = dst[e];
  const float* q  = Q + (size_t)d * HD + h * DHEAD;
  const float* kk = K + (size_t)s * HD + h * DHEAD;
  float acc = 0.f;
#pragma unroll
  for (int i = 0; i < DHEAD; ++i) acc += q[i] * kk[i];
  float logit = acc * 0.17677669529663687f;   // 1/sqrt(32)
  elog[idx] = logit;
  atomicMax(&mx[(size_t)d * NHEAD + h], f2ord(logit));
}

__global__ void gtr_edge_exp(float* __restrict__ elog, const unsigned* __restrict__ mx,
                             float* __restrict__ dn, const int* __restrict__ dst, long Ne) {
  long idx = (long)blockIdx.x * blockDim.x + threadIdx.x;
  if (idx >= Ne * NHEAD) return;
  long e = idx >> 2; int h = (int)(idx & 3);
  int d = dst[e];
  float m = ord2f(mx[(size_t)d * NHEAD + h]);
  float ex = __expf(elog[idx] - m);
  elog[idx] = ex;
  atomicAdd(&dn[(size_t)d * NHEAD + h], ex);
}

__global__ void gtr_edge_scatter(const float* __restrict__ elog, const float* __restrict__ dn,
                                 const float* __restrict__ V, const int* __restrict__ src,
                                 const int* __restrict__ dst, float* __restrict__ Om, long Ne) {
  long idx = (long)blockIdx.x * blockDim.x + threadIdx.x;
  if (idx >= Ne * HD) return;
  long e = idx >> 7; int c = (int)(idx & 127); int h = c >> 5;
  int s = src[e], d = dst[e];
  float alpha = elog[e * NHEAD + h] / (dn[(size_t)d * NHEAD + h] + 1e-16f);
  atomicAdd(&Om[(size_t)d * HD + c], alpha * V[(size_t)s * HD + c]);
}

// --------------------- beta-gated skip + relu (wave per node) ---------------
__global__ void gtr_beta(const float* __restrict__ Om, const float* __restrict__ Rm,
                         const float* __restrict__ Wb, float* __restrict__ Hout,
                         _Float16* __restrict__ hh, int Nn) {
  int node = blockIdx.x * 8 + (threadIdx.x >> 5);
  int lane = threadIdx.x & 31;
  if (node >= Nn) return;
  const float* o = Om + (size_t)node * HD;
  const float* r = Rm + (size_t)node * HD;
  float s = 0.f;
  for (int j = lane; j < HD; j += 32) {
    float ov = o[j], rv = r[j];
    s += ov * Wb[j] + rv * Wb[HD + j] + (ov - rv) * Wb[2 * HD + j];
  }
#pragma unroll
  for (int off = 16; off; off >>= 1) s += __shfl_xor(s, off, 32);
  float beta = 1.f / (1.f + __expf(-s));
  for (int j = lane; j < HD; j += 32) {
    float v = beta * r[j] + (1.f - beta) * o[j];
    v = v > 0.f ? v : 0.f;
    Hout[(size_t)node * HD + j] = v;
    hh[(size_t)node * HD + j]   = (_Float16)v;
  }
}

// ------------------------------- pooling -----------------------------------
__global__ void gtr_pool(const float* __restrict__ H, const int* __restrict__ batch,
                         float* __restrict__ pooled, float* __restrict__ cnt, int Nn) {
  long i = (long)blockIdx.x * blockDim.x + threadIdx.x;
  long tot = (long)Nn * HD;
  if (i >= tot) return;
  int n = (int)(i >> 7); int j = (int)(i & 127);
  int g = batch[n];
  atomicAdd(&pooled[(size_t)g * HD + j], H[i]);
  if (j == 0) atomicAdd(&cnt[g], 1.f);
}

__global__ void gtr_final(const float* __restrict__ pooled, const float* __restrict__ cnt,
                          const float* __restrict__ Wfc, float* __restrict__ out, int Gg) {
  int g = blockIdx.x * 8 + (threadIdx.x >> 5);
  int lane = threadIdx.x & 31;
  if (g >= Gg) return;
  float s = 0.f;
  for (int j = lane; j < HD; j += 32) s += pooled[(size_t)g * HD + j] * Wfc[j];
#pragma unroll
  for (int off = 16; off; off >>= 1) s += __shfl_xor(s, off, 32);
  if (lane == 0) {
    float c = cnt[g]; if (c < 1.f) c = 1.f;
    out[g] = 1.f / (1.f + __expf(-s / c));
  }
}

// ---------------------------------------------------------------------------
extern "C" void kernel_launch(void* const* d_in, const int* in_sizes, int n_in,
                              void* d_out, int out_size, void* d_ws, size_t ws_size,
                              hipStream_t stream) {
  const int  Nn = in_sizes[0] / 4;     // 100000 (x is N x 4)
  const long Ne = in_sizes[1] / 2;     // 800000 (edge_index is 2 x E)
  const int  Gg = out_size;            // 64 graphs

  const float* x     = (const float*)d_in[0];
  const int*   srcI  = (const int*)d_in[1];        // row 0 of edge_index
  const int*   dstI  = srcI + Ne;                  // row 1 of edge_index
  const int*   batch = (const int*)d_in[2];
  // params: per layer [Wq,bq,Wk,bk,Wv,bv,Ws,bs,Wb] (insertion order), 3 layers
  const float* Wfc   = (const float*)d_in[3 + 3 * 9];

  char* ws = (char*)d_ws;
  size_t off = 0;
  auto carve = [&](size_t bytes) -> char* {
    char* p = ws + off; off = (off + bytes + 255) & ~(size_t)255; return p;
  };
  _Float16* hh  = (_Float16*)carve((size_t)Nn * HD * sizeof(_Float16));
  float* Qm     = (float*)carve((size_t)Nn * HD * sizeof(float));
  float* Km     = (float*)carve((size_t)Nn * HD * sizeof(float));
  float* Vm     = (float*)carve((size_t)Nn * HD * sizeof(float));
  float* Rm     = (float*)carve((size_t)Nn * HD * sizeof(float));
  float* Om     = (float*)carve((size_t)Nn * HD * sizeof(float));
  float* Hc     = (float*)carve((size_t)Nn * HD * sizeof(float));
  _Float16* Wh  = (_Float16*)carve((size_t)4 * HD * HD * sizeof(_Float16));
  float* elog   = (float*)carve((size_t)Ne * NHEAD * sizeof(float));
  unsigned* mx  = (unsigned*)carve((size_t)Nn * NHEAD * sizeof(unsigned));
  float* dn     = (float*)carve((size_t)Nn * NHEAD * sizeof(float));
  float* pooled = (float*)carve((size_t)Gg * HD * sizeof(float));
  float* cnt    = (float*)carve((size_t)Gg * sizeof(float));

  const int TPB = 256;
  auto blks = [](long n, int t) { return (unsigned)((n + t - 1) / t); };

  const int Mtiles = (Nn + 15) / 16;
  const unsigned gemm_blocks = (unsigned)((Mtiles + MT_PER_BLK - 1) / MT_PER_BLK);

  for (int l = 0; l < 3; ++l) {
    const int cin = (l == 0) ? 4 : HD;
    const int Kd  = (l == 0) ? 32 : HD;     // layer-1 K zero-padded to 32
    const float* Wq = (const float*)d_in[3 + l * 9 + 0];
    const float* bq = (const float*)d_in[3 + l * 9 + 1];
    const float* Wk = (const float*)d_in[3 + l * 9 + 2];
    const float* bk = (const float*)d_in[3 + l * 9 + 3];
    const float* Wv = (const float*)d_in[3 + l * 9 + 4];
    const float* bv = (const float*)d_in[3 + l * 9 + 5];
    const float* Ws = (const float*)d_in[3 + l * 9 + 6];
    const float* bs = (const float*)d_in[3 + l * 9 + 7];
    const float* Wb = (const float*)d_in[3 + l * 9 + 8];

    if (l == 0) {
      long tot = (long)Nn * 32;
      gtr_cast_pad<<<blks(tot, TPB), TPB, 0, stream>>>(x, hh, 4, 32, Nn);
    }
    int wtot = Kd * HD;
    gtr_castw<<<blks(wtot, TPB), TPB, 0, stream>>>(Wq, Wh + 0 * HD * HD, cin, Kd);
    gtr_castw<<<blks(wtot, TPB), TPB, 0, stream>>>(Wk, Wh + 1 * HD * HD, cin, Kd);
    gtr_castw<<<blks(wtot, TPB), TPB, 0, stream>>>(Wv, Wh + 2 * HD * HD, cin, Kd);
    gtr_castw<<<blks(wtot, TPB), TPB, 0, stream>>>(Ws, Wh + 3 * HD * HD, cin, Kd);

    if (l == 0) {
      gtr_gemm_wmma<32><<<gemm_blocks, 256, 0, stream>>>(hh, Wh + 0 * HD * HD, bq, Qm, Nn);
      gtr_gemm_wmma<32><<<gemm_blocks, 256, 0, stream>>>(hh, Wh + 1 * HD * HD, bk, Km, Nn);
      gtr_gemm_wmma<32><<<gemm_blocks, 256, 0, stream>>>(hh, Wh + 2 * HD * HD, bv, Vm, Nn);
      gtr_gemm_wmma<32><<<gemm_blocks, 256, 0, stream>>>(hh, Wh + 3 * HD * HD, bs, Rm, Nn);
    } else {
      gtr_gemm_wmma<128><<<gemm_blocks, 256, 0, stream>>>(hh, Wh + 0 * HD * HD, bq, Qm, Nn);
      gtr_gemm_wmma<128><<<gemm_blocks, 256, 0, stream>>>(hh, Wh + 1 * HD * HD, bk, Km, Nn);
      gtr_gemm_wmma<128><<<gemm_blocks, 256, 0, stream>>>(hh, Wh + 2 * HD * HD, bv, Vm, Nn);
      gtr_gemm_wmma<128><<<gemm_blocks, 256, 0, stream>>>(hh, Wh + 3 * HD * HD, bs, Rm, Nn);
    }

    gtr_fill_f32<<<blks((long)Nn * HD, TPB), TPB, 0, stream>>>(Om, 0.f, (long)Nn * HD);
    gtr_fill_u32<<<blks((long)Nn * NHEAD, TPB), TPB, 0, stream>>>(mx, ORD_NEG_INF, (long)Nn * NHEAD);
    gtr_fill_f32<<<blks((long)Nn * NHEAD, TPB), TPB, 0, stream>>>(dn, 0.f, (long)Nn * NHEAD);

    gtr_edge_logits<<<blks(Ne * NHEAD, TPB), TPB, 0, stream>>>(Qm, Km, srcI, dstI, elog, mx, Ne);
    gtr_edge_exp<<<blks(Ne * NHEAD, TPB), TPB, 0, stream>>>(elog, mx, dn, dstI, Ne);
    gtr_edge_scatter<<<blks(Ne * HD, TPB), TPB, 0, stream>>>(elog, dn, Vm, srcI, dstI, Om, Ne);

    gtr_beta<<<blks(Nn, 8), 256, 0, stream>>>(Om, Rm, Wb, Hc, hh, Nn);
  }

  gtr_fill_f32<<<blks((long)Gg * HD, TPB), TPB, 0, stream>>>(pooled, 0.f, (long)Gg * HD);
  gtr_fill_f32<<<1, 64, 0, stream>>>(cnt, 0.f, Gg);
  gtr_pool<<<blks((long)Nn * HD, TPB), TPB, 0, stream>>>(Hc, batch, pooled, cnt, Nn);
  gtr_final<<<blks(Gg, 8), 256, 0, stream>>>(pooled, cnt, Wfc, (float*)d_out, Gg);
}
